// KNNFeatureProcessor_55070070670321
// MI455X (gfx1250) — matrix operand order repaired and checked by reading, not compiled, not run on hardware
//
#include <hip/hip_runtime.h>

// ---------------------------------------------------------------------------
// KNN feature processor for MI455X (gfx1250), wave32, bf16 WMMA everywhere.
// B=65536, D=256, BANK=1000 (padded to 1008), K=5, TEMP=1.
// M=32 rows per block: each B-fragment load feeds two WMMA chains, halving
// L2 traffic for the bank / weight operands (the dominant cost).
// ---------------------------------------------------------------------------

typedef __attribute__((ext_vector_type(16))) __bf16 v16bf;
typedef __attribute__((ext_vector_type(8)))  float  v8f;

union AF {
    v16bf    v;
    uint4    q[2];
    unsigned u[8];
};

// round-to-nearest-even f32 -> bf16 (bits)
__device__ __forceinline__ unsigned short f2bf(float f) {
    unsigned u = __float_as_uint(f);
    unsigned r = u + 0x7FFFu + ((u >> 16) & 1u);
    return (unsigned short)(r >> 16);
}

__device__ __forceinline__ unsigned pk2(float x, float y) {
    return (unsigned)f2bf(x) | ((unsigned)f2bf(y) << 16);
}

// ---------------------------------------------------------------------------
// Kernel 0: convert W1 [256,512] and W2 [256,256] (torch [out,in]) to bf16.
// ---------------------------------------------------------------------------
__global__ void convert_weights(const float* __restrict__ W1,
                                const float* __restrict__ W2,
                                unsigned short* __restrict__ w1b,
                                unsigned short* __restrict__ w2b) {
    int i = blockIdx.x * 256 + threadIdx.x;
    if (i < 256 * 512) {
        w1b[i] = f2bf(W1[i]);
    } else {
        int j = i - 256 * 512;
        if (j < 256 * 256) w2b[j] = f2bf(W2[j]);
    }
}

// ---------------------------------------------------------------------------
// Kernel 1: L2-normalize bank rows -> bf16, pad rows [1000,1008) with zeros.
// One block (256 threads) per row.
// ---------------------------------------------------------------------------
__global__ void normalize_bank(const float* __restrict__ bank,
                               unsigned short* __restrict__ bankb) {
    const int r = blockIdx.x;
    const int t = threadIdx.x;
    float v = (r < 1000) ? bank[r * 256 + t] : 0.0f;
    float s = v * v;
    s += __shfl_xor(s, 1, 32);  s += __shfl_xor(s, 2, 32);
    s += __shfl_xor(s, 4, 32);  s += __shfl_xor(s, 8, 32);
    s += __shfl_xor(s, 16, 32);
    __shared__ float partial[8];
    __shared__ float scale;
    if ((t & 31) == 0) partial[t >> 5] = s;
    __syncthreads();
    if (t == 0) {
        float tot = 0.0f;
        #pragma unroll
        for (int i = 0; i < 8; ++i) tot += partial[i];
        scale = 1.0f / fmaxf(sqrtf(tot), 1e-12f);
    }
    __syncthreads();
    bankb[r * 256 + t] = f2bf(v * scale);
}

// ---------------------------------------------------------------------------
// Kernel 2: per 32-row tile: sims via bf16 WMMA vs all 1008 (padded) bank
// rows, top-5 + softmax + weighted neighbor average -> nf (bf16).
// 256 threads = 8 waves per block; 2048 blocks. Two M-tiles share each
// B fragment (bank rows) -> half the L2 traffic of M=16 blocking.
// ---------------------------------------------------------------------------
__global__ void __launch_bounds__(256)
sims_topk_nf(const float* __restrict__ feats,
             const float* __restrict__ bank,
             const unsigned short* __restrict__ bankb,
             unsigned short* __restrict__ nfb) {
    __shared__ float sims[32 * 1008];   // 129024 B (CDNA5: up to 320KB/WG)
    __shared__ float rnorm[32];

    const int  t       = threadIdx.x;
    const int  lane    = t & 31;
    const int  wave    = t >> 5;
    const int  m16     = lane & 15;
    const int  half    = lane >> 4;
    const long rowBase = (long)blockIdx.x * 32;

    // ---- per-row inverse L2 norms (8 threads per row, sub-wave reduce) ----
    {
        const int row = t >> 3, g = t & 7;
        const float* fr = feats + (rowBase + row) * 256;
        float s = 0.0f;
        #pragma unroll
        for (int j = 0; j < 32; ++j) { float x = fr[g * 32 + j]; s += x * x; }
        s += __shfl_xor(s, 1, 32);  s += __shfl_xor(s, 2, 32);
        s += __shfl_xor(s, 4, 32);
        if (g == 0) rnorm[row] = 1.0f / fmaxf(sqrtf(s), 1e-12f);
    }
    __syncthreads();

    // ---- build A-fragments for both M-tiles (normalized queries, bf16) ----
    // 16-bit A layout: lanes 0-15 hold K {0..7,16..23}, lanes 16-31 {8..15,24..31}
    AF a0[8], a1[8];
    #pragma unroll
    for (int mt = 0; mt < 2; ++mt) {
        const int    row = mt * 16 + m16;
        const float* fr  = feats + (rowBase + row) * 256;
        const float  rn  = rnorm[row];
        AF* a = mt ? a1 : a0;
        #pragma unroll
        for (int kc = 0; kc < 8; ++kc) {
            const int b0 = kc * 32 + half * 8;
            const float4* p = (const float4*)(fr + b0);
            const float4* q = (const float4*)(fr + b0 + 16);
            float4 x0 = p[0], x1 = p[1], y0 = q[0], y1 = q[1];
            a[kc].u[0] = pk2(x0.x * rn, x0.y * rn);
            a[kc].u[1] = pk2(x0.z * rn, x0.w * rn);
            a[kc].u[2] = pk2(x1.x * rn, x1.y * rn);
            a[kc].u[3] = pk2(x1.z * rn, x1.w * rn);
            a[kc].u[4] = pk2(y0.x * rn, y0.y * rn);
            a[kc].u[5] = pk2(y0.z * rn, y0.w * rn);
            a[kc].u[6] = pk2(y1.x * rn, y1.y * rn);
            a[kc].u[7] = pk2(y1.z * rn, y1.w * rn);
        }
    }

    // ---- sims: each wave sweeps N-tiles {wave, wave+8, ...} of 63 ----
    for (int tile = wave; tile < 63; tile += 8) {
        const int n0 = tile * 16;
        v8f acc0 = {};
        v8f acc1 = {};
        #pragma unroll
        for (int kc = 0; kc < 8; ++kc) {
            AF b;
            const uint4* pb =
                (const uint4*)(bankb + (n0 + m16) * 256 + kc * 32 + half * 8);
            b.q[0] = pb[0];
            b.q[1] = pb[2];
            acc0 = __builtin_amdgcn_wmma_f32_16x16x32_bf16(
                false, a0[kc].v, false, b.v, (short)0, acc0, false, false);
            acc1 = __builtin_amdgcn_wmma_f32_16x16x32_bf16(
                false, a1[kc].v, false, b.v, (short)0, acc1, false, false);
        }
        const int col = n0 + m16;
        #pragma unroll
        for (int e = 0; e < 8; ++e) {
            sims[(e + 8 * half) * 1008 + col]      = (col < 1000) ? acc0[e] : -3.4e38f;
            sims[(16 + e + 8 * half) * 1008 + col] = (col < 1000) ? acc1[e] : -3.4e38f;
        }
    }
    __syncthreads();

    // ---- top-5 / softmax / neighbor average: 4 rows per wave ----
    for (int rr = 0; rr < 4; ++rr) {
        const int r = wave * 4 + rr;
        float bv0 = -3.4e38f, bv1 = -3.4e38f, bv2 = -3.4e38f,
              bv3 = -3.4e38f, bv4 = -3.4e38f;
        int   bi0 = -1, bi1 = -1, bi2 = -1, bi3 = -1, bi4 = -1;
        for (int c = lane; c < 1008; c += 32) {
            const float x = sims[r * 1008 + c];
            if (x > bv4) {
                if (x > bv0)      { bv4=bv3;bi4=bi3; bv3=bv2;bi3=bi2; bv2=bv1;bi2=bi1; bv1=bv0;bi1=bi0; bv0=x;bi0=c; }
                else if (x > bv1) { bv4=bv3;bi4=bi3; bv3=bv2;bi3=bi2; bv2=bv1;bi2=bi1; bv1=x;bi1=c; }
                else if (x > bv2) { bv4=bv3;bi4=bi3; bv3=bv2;bi3=bi2; bv2=x;bi2=c; }
                else if (x > bv3) { bv4=bv3;bi4=bi3; bv3=x;bi3=c; }
                else              { bv4=x;bi4=c; }
            }
        }
        float tv[5]; int ti[5];
        #pragma unroll
        for (int s = 0; s < 5; ++s) {
            float x = bv0; int id = bi0;
            #pragma unroll
            for (int off = 16; off; off >>= 1) {
                float ox = __shfl_xor(x, off, 32);
                int   oi = __shfl_xor(id, off, 32);
                if (ox > x || (ox == x && oi < id)) { x = ox; id = oi; }
            }
            tv[s] = x; ti[s] = id;
            if (bi0 == id) {  // winning lane pops its head (indices unique)
                bv0=bv1;bi0=bi1; bv1=bv2;bi1=bi2; bv2=bv3;bi2=bi3;
                bv3=bv4;bi3=bi4; bv4=-3.4e38f;bi4=-1;
            }
        }
        // softmax (TEMP = 1, tv[0] is the max)
        float wk[5], sum = 0.0f;
        #pragma unroll
        for (int s = 0; s < 5; ++s) { wk[s] = __expf(tv[s] - tv[0]); sum += wk[s]; }
        const float inv = 1.0f / sum;
        // weighted average of original (unnormalized) bank rows; 8 dims/lane
        const int d0 = lane * 8;
        float nf[8] = {0, 0, 0, 0, 0, 0, 0, 0};
        #pragma unroll
        for (int s = 0; s < 5; ++s) {
            const float w = wk[s] * inv;
            const float4* bp = (const float4*)(bank + (long)ti[s] * 256 + d0);
            float4 p0 = bp[0], p1 = bp[1];
            nf[0] += w * p0.x; nf[1] += w * p0.y; nf[2] += w * p0.z; nf[3] += w * p0.w;
            nf[4] += w * p1.x; nf[5] += w * p1.y; nf[6] += w * p1.z; nf[7] += w * p1.w;
        }
        unsigned short* np = nfb + (rowBase + r) * 256 + d0;
        #pragma unroll
        for (int j = 0; j < 8; ++j) np[j] = f2bf(nf[j]);
    }
}

// ---------------------------------------------------------------------------
// Kernel 3: fusion MLP. h = relu([feat|nf] @ W1^T + b1); out = h @ W2^T + b2.
// 256 threads = 8 waves, 32 rows per block; both GEMMs via bf16 WMMA with
// two M-tiles sharing each weight fragment.
// ---------------------------------------------------------------------------
__global__ void __launch_bounds__(256)
fusion_mlp(const float* __restrict__ feats,
           const unsigned short* __restrict__ nfb,
           const unsigned short* __restrict__ w1b,
           const float* __restrict__ b1,
           const unsigned short* __restrict__ w2b,
           const float* __restrict__ b2,
           float* __restrict__ out) {
    __shared__ unsigned short fused[32 * 512];  // 32 KB
    __shared__ unsigned short hbuf[32 * 256];   // 16 KB

    const int  t       = threadIdx.x;
    const int  lane    = t & 31;
    const int  wave    = t >> 5;
    const int  m16     = lane & 15;
    const int  half    = lane >> 4;
    const long rowBase = (long)blockIdx.x * 32;

    // ---- stage fused = [features(bf16) | nf(bf16)] row-major [32][512] ----
    {
        const int row = t >> 3, g = t & 7;  // 8 threads per row, 64 elems each
        unsigned short* dst = fused + row * 512;
        const float*          fr = feats + (rowBase + row) * 256;
        const unsigned short* nr = nfb   + (rowBase + row) * 256;
        const int d = g * 64;
        if (d < 256) {
            #pragma unroll
            for (int j = 0; j < 64; ++j) dst[d + j] = f2bf(fr[d + j]);
        } else {
            #pragma unroll
            for (int j = 0; j < 64; ++j) dst[d + j] = nr[d - 256 + j];
        }
    }
    __syncthreads();

    // ---- layer 1: K = 512 (16 WMMA / M-tile), N = 256 -> 2 tiles per wave ----
    #pragma unroll
    for (int tt = 0; tt < 2; ++tt) {
        const int n0 = (wave + 8 * tt) * 16;
        v8f acc0 = {};
        v8f acc1 = {};
        for (int kc = 0; kc < 16; ++kc) {
            const int k0 = kc * 32;
            AF af0, af1, bfr;
            af0.q[0] = *(const uint4*)(fused + m16 * 512 + k0 + half * 8);
            af0.q[1] = *(const uint4*)(fused + m16 * 512 + k0 + 16 + half * 8);
            af1.q[0] = *(const uint4*)(fused + (16 + m16) * 512 + k0 + half * 8);
            af1.q[1] = *(const uint4*)(fused + (16 + m16) * 512 + k0 + 16 + half * 8);
            const uint4* pb =
                (const uint4*)(w1b + (n0 + m16) * 512 + k0 + half * 8);
            bfr.q[0] = pb[0];
            bfr.q[1] = pb[2];
            acc0 = __builtin_amdgcn_wmma_f32_16x16x32_bf16(
                false, af0.v, false, bfr.v, (short)0, acc0, false, false);
            acc1 = __builtin_amdgcn_wmma_f32_16x16x32_bf16(
                false, af1.v, false, bfr.v, (short)0, acc1, false, false);
        }
        const int   col  = n0 + m16;
        const float bias = b1[col];
        #pragma unroll
        for (int e = 0; e < 8; ++e) {
            hbuf[(e + 8 * half) * 256 + col]      = f2bf(fmaxf(acc0[e] + bias, 0.0f));
            hbuf[(16 + e + 8 * half) * 256 + col] = f2bf(fmaxf(acc1[e] + bias, 0.0f));
        }
    }
    __syncthreads();

    // ---- layer 2: K = 256 (8 WMMA / M-tile), N = 256 -> 2 tiles per wave ----
    #pragma unroll
    for (int tt = 0; tt < 2; ++tt) {
        const int n0 = (wave + 8 * tt) * 16;
        v8f acc0 = {};
        v8f acc1 = {};
        #pragma unroll
        for (int kc = 0; kc < 8; ++kc) {
            const int k0 = kc * 32;
            AF af0, af1, bfr;
            af0.q[0] = *(const uint4*)(hbuf + m16 * 256 + k0 + half * 8);
            af0.q[1] = *(const uint4*)(hbuf + m16 * 256 + k0 + 16 + half * 8);
            af1.q[0] = *(const uint4*)(hbuf + (16 + m16) * 256 + k0 + half * 8);
            af1.q[1] = *(const uint4*)(hbuf + (16 + m16) * 256 + k0 + 16 + half * 8);
            const uint4* pb =
                (const uint4*)(w2b + (n0 + m16) * 256 + k0 + half * 8);
            bfr.q[0] = pb[0];
            bfr.q[1] = pb[2];
            acc0 = __builtin_amdgcn_wmma_f32_16x16x32_bf16(
                false, af0.v, false, bfr.v, (short)0, acc0, false, false);
            acc1 = __builtin_amdgcn_wmma_f32_16x16x32_bf16(
                false, af1.v, false, bfr.v, (short)0, acc1, false, false);
        }
        const int   col  = n0 + m16;
        const float bias = b2[col];
        #pragma unroll
        for (int e = 0; e < 8; ++e) {
            out[(rowBase + e + 8 * half) * 256 + col]      = acc0[e] + bias;
            out[(rowBase + 16 + e + 8 * half) * 256 + col] = acc1[e] + bias;
        }
    }
}

// ---------------------------------------------------------------------------
// Launch
// ---------------------------------------------------------------------------
extern "C" void kernel_launch(void* const* d_in, const int* in_sizes, int n_in,
                              void* d_out, int out_size, void* d_ws, size_t ws_size,
                              hipStream_t stream) {
    const float* feats = (const float*)d_in[0];   // [65536, 256]
    const float* bank  = (const float*)d_in[1];   // [1000, 256]
    const float* W1    = (const float*)d_in[2];   // [256, 512]
    const float* b1    = (const float*)d_in[3];   // [256]
    const float* W2    = (const float*)d_in[4];   // [256, 256]
    const float* b2    = (const float*)d_in[5];   // [256]
    float* out = (float*)d_out;                   // [65536, 256]

    char* ws = (char*)d_ws;
    // all offsets 256-byte aligned
    unsigned short* bankb = (unsigned short*)(ws);                              // 1008*256*2 = 516096
    unsigned short* w1b   = (unsigned short*)(ws + 516096);                     // 256*512*2  = 262144
    unsigned short* w2b   = (unsigned short*)(ws + 516096 + 262144);            // 256*256*2  = 131072
    unsigned short* nfb   = (unsigned short*)(ws + 516096 + 262144 + 131072);   // 65536*256*2

    hipLaunchKernelGGL(convert_weights, dim3(768),  dim3(256), 0, stream, W1, W2, w1b, w2b);
    hipLaunchKernelGGL(normalize_bank,  dim3(1008), dim3(256), 0, stream, bank, bankb);
    hipLaunchKernelGGL(sims_topk_nf,    dim3(2048), dim3(256), 0, stream, feats, bank, bankb, nfb);
    hipLaunchKernelGGL(fusion_mlp,      dim3(2048), dim3(256), 0, stream, feats, nfb, w1b, b1, w2b, b2, out);
}
